// Inference_Sinkhorn_Loss_13357348290934
// MI455X (gfx1250) — compile-verified, matrix operand order checked by reading, not run on hardware
//
#include <hip/hip_runtime.h>

#define NITER 10
#define CDIM  128
#define TEMP  0.07f

typedef float vf4 __attribute__((ext_vector_type(4)));
typedef __attribute__((ext_vector_type(16))) _Float16 v16h;
typedef __attribute__((ext_vector_type(8)))  float    v8f;

// ---------------------------------------------------------------------------
// init: b[c] = ratios[c] * total_num ; v[c] = 1
// ---------------------------------------------------------------------------
__global__ void init_bv(const float* __restrict__ ratios,
                        const int* __restrict__ totalNum,
                        float* __restrict__ bvec, float* __restrict__ v) {
  int c = threadIdx.x;
  bvec[c] = ratios[c] * (float)(*totalNum);
  v[c]    = 1.0f;
}

// ---------------------------------------------------------------------------
// Fused Sinkhorn sweep. One wave (32 lanes) owns a row: lane j holds columns
// 4j..4j+3 (one 512B coalesced float4 load per row per wave).
//   rowdot -> butterfly shfl reduce -> u[i] = 1/rowdot (all lanes have it)
//   column partials K[i,c]*u[i] accumulate in registers, reduced across the
//   8 waves of the block through LDS in fixed order, written per-block.
// Rows < nres use regular-temporal loads (L2-resident ~154MB working set);
// rows >= nres use non-temporal loads so the stream doesn't thrash L2.
// ---------------------------------------------------------------------------
__global__ __launch_bounds__(256) void sinkhorn_pass(
    const float* __restrict__ S, float* __restrict__ u,
    const float* __restrict__ v, float* __restrict__ partial,
    int N, int nres) {
  __shared__ float scd[8 * CDIM];
  const int lane = threadIdx.x & 31;
  const int wave = threadIdx.x >> 5;
  const vf4 v4 = ((const vf4*)v)[lane];
  float a0 = 0.f, a1 = 0.f, a2 = 0.f, a3 = 0.f;
  const int gw = blockIdx.x * 8 + wave;
  const int tw = (int)gridDim.x * 8;
  for (int row = gw; row < N; row += tw) {
    const vf4* p = (const vf4*)(S + (size_t)row * CDIM);
    __builtin_prefetch(S + ((size_t)row + (size_t)tw) * CDIM + (size_t)lane * 4, 0, 0);
    vf4 s4 = (row < nres) ? p[lane] : __builtin_nontemporal_load(p + lane);
    float e0 = __expf(s4.x), e1 = __expf(s4.y), e2 = __expf(s4.z), e3 = __expf(s4.w);
    float dot = e0 * v4.x + e1 * v4.y + e2 * v4.z + e3 * v4.w;
#pragma unroll
    for (int off = 16; off > 0; off >>= 1) dot += __shfl_xor(dot, off, 32);
    float ui = 1.0f / dot;           // valid in every lane after butterfly
    if (lane == 0) u[row] = ui;
    a0 += e0 * ui; a1 += e1 * ui; a2 += e2 * ui; a3 += e3 * ui;
  }
  scd[wave * CDIM + 4 * lane + 0] = a0;
  scd[wave * CDIM + 4 * lane + 1] = a1;
  scd[wave * CDIM + 4 * lane + 2] = a2;
  scd[wave * CDIM + 4 * lane + 3] = a3;
  __syncthreads();
  if (threadIdx.x < CDIM) {
    float s = 0.f;
#pragma unroll
    for (int w = 0; w < 8; ++w) s += scd[w * CDIM + threadIdx.x];
    partial[(size_t)blockIdx.x * CDIM + threadIdx.x] = s;   // no atomics
  }
}

// ---------------------------------------------------------------------------
// Deterministic column reduction of per-block partials; v[c] = b[c]/coldot[c]
// One block per column, fixed-order strided + tree reduce.
// ---------------------------------------------------------------------------
__global__ __launch_bounds__(256) void update_v(
    const float* __restrict__ partial, const float* __restrict__ bvec,
    float* __restrict__ v, int G) {
  __shared__ float red[256];
  const int c = blockIdx.x;
  float s = 0.f;
  for (int i = threadIdx.x; i < G; i += 256) s += partial[(size_t)i * CDIM + c];
  red[threadIdx.x] = s;
  __syncthreads();
  for (int st = 128; st > 0; st >>= 1) {
    if (threadIdx.x < st) red[threadIdx.x] += red[threadIdx.x + st];
    __syncthreads();
  }
  if (threadIdx.x == 0) v[c] = bvec[c] / red[0];
}

// ---------------------------------------------------------------------------
// P[i,c] = exp(S[i,c]) * u[i] * v[c]   (non-temporal stores: never re-read)
// ---------------------------------------------------------------------------
__global__ __launch_bounds__(256) void finalize_p(
    const float* __restrict__ S, const float* __restrict__ u,
    const float* __restrict__ v, float* __restrict__ out, int N, int nres) {
  const size_t total  = (size_t)N * 32;              // float4 chunks
  const size_t stride = (size_t)gridDim.x * 256;
  for (size_t idx = (size_t)blockIdx.x * 256 + threadIdx.x; idx < total; idx += stride) {
    const size_t row = idx >> 5;
    const int part = (int)(idx & 31);
    const vf4* sp = (const vf4*)S + idx;
    vf4 s4 = ((long long)row < (long long)nres) ? *sp : __builtin_nontemporal_load(sp);
    vf4 v4 = ((const vf4*)v)[part];
    float ui = u[row];
    vf4 r;
    r.x = __expf(s4.x) * ui * v4.x;
    r.y = __expf(s4.y) * ui * v4.y;
    r.z = __expf(s4.z) * ui * v4.z;
    r.w = __expf(s4.w) * ui * v4.w;
    __builtin_nontemporal_store(r, (vf4*)out + idx);
  }
}

// ---------------------------------------------------------------------------
// sim = features @ text^T / TEMP via v_wmma_f32_16x16x32_f16 (reference
// computes this; value unused downstream -> goes to workspace).
// One wave per 16x16 output tile; block = one M-tile row (8 N-tiles).
// A fragment per ISA 16-bit 16x32 layout: lane L -> M=L%16,
// elems 0..7 = K kb+(L<16?0:8)+j, elems 8..15 = K kb+16+(L<16?0:8)+j.
// ---------------------------------------------------------------------------
__global__ __launch_bounds__(256) void simmat_wmma(
    const float* __restrict__ A, const float* __restrict__ Bm,
    float* __restrict__ out) {
  const int lane  = threadIdx.x & 31;
  const int tn    = threadIdx.x >> 5;        // 0..7  (128/16 N-tiles)
  const int tm    = blockIdx.x;              // M-tile
  const int m     = lane & 15;
  const int khalf = (lane >> 4) * 8;
  const float* arow = A  + (size_t)(tm * 16 + m) * 512;
  const float* brow = Bm + (size_t)(tn * 16 + m) * 512;
  v8f acc = {};
  for (int kb = 0; kb < 512; kb += 32) {
    v16h a, b;
#pragma unroll
    for (int j = 0; j < 8; ++j) {
      a[j]     = (_Float16)arow[kb + khalf + j];
      a[j + 8] = (_Float16)arow[kb + 16 + khalf + j];
      b[j]     = (_Float16)brow[kb + khalf + j];
      b[j + 8] = (_Float16)brow[kb + 16 + khalf + j];
    }
    acc = __builtin_amdgcn_wmma_f32_16x16x32_f16(
        false, a, false, b, (short)0, acc, false, false);
  }
  const float invT = 1.0f / TEMP;
  const int n = lane & 15;
  const int mbase = (lane >> 4) * 8;
#pragma unroll
  for (int r = 0; r < 8; ++r)
    out[(size_t)(tm * 16 + mbase + r) * CDIM + tn * 16 + n] = acc[r] * invT;
}

// ---------------------------------------------------------------------------
extern "C" void kernel_launch(void* const* d_in, const int* in_sizes, int n_in,
                              void* d_out, int out_size, void* d_ws, size_t ws_size,
                              hipStream_t stream) {
  const float* features = (const float*)d_in[0];
  const float* textf    = (const float*)d_in[1];
  const float* S        = (const float*)d_in[2];
  const float* ratios   = (const float*)d_in[3];
  const int*   totalNum = (const int*)d_in[5];

  const int N = in_sizes[2] / CDIM;          // 500000
  float* out = (float*)d_out;

  const int G = 2000;                        // pass-kernel blocks (16000 waves)
  // L2-resident split: keep ~154MB of S regular-temporal, stream the rest NT.
  int nres = N < 300000 ? N : 300000;

  // workspace layout
  char* ws = (char*)d_ws;
  size_t offU = 0;                                   // u: N floats
  size_t offB = (((size_t)N * 4) + 255) & ~(size_t)255;
  size_t offV   = offB + 512;
  size_t offP   = offV + 512;                        // partials: G*128 floats
  size_t offSim = offP + (size_t)G * CDIM * 4;
  float* u       = (float*)(ws + offU);
  float* bvec    = (float*)(ws + offB);
  float* v       = (float*)(ws + offV);
  float* partial = (float*)(ws + offP);

  init_bv<<<1, CDIM, 0, stream>>>(ratios, totalNum, bvec, v);
  for (int it = 0; it < NITER; ++it) {
    sinkhorn_pass<<<G, 256, 0, stream>>>(S, u, v, partial, N, nres);
    update_v<<<CDIM, 256, 0, stream>>>(partial, bvec, v, G);
  }
  finalize_p<<<4096, 256, 0, stream>>>(S, u, v, out, N, nres);

  // Dead sim_matrix the reference forward also computes (WMMA path).
  int D    = in_sizes[1] / CDIM;                     // 512
  int Brow = (D > 0) ? in_sizes[0] / D : 0;          // 4096
  if (D == 512 && Brow > 0 && (Brow % 16) == 0 &&
      ws_size >= offSim + (size_t)Brow * CDIM * 4) {
    float* sim = (float*)(ws + offSim);
    simmat_wmma<<<Brow / 16, 256, 0, stream>>>(features, textf, sim);
  }
}
